// UNet4DAntoine2_38912403701994
// MI455X (gfx1250) — matrix operand order repaired and checked by reading, not compiled, not run on hardware
//
#include <hip/hip_runtime.h>

// ---------------------------------------------------------------------------
// Types / helpers
// ---------------------------------------------------------------------------
typedef __attribute__((ext_vector_type(16))) __bf16        v16bf;
typedef __attribute__((ext_vector_type(8)))  float         v8f;
typedef __attribute__((ext_vector_type(4)))  unsigned int  u32x4;
typedef __attribute__((ext_vector_type(2)))  unsigned int  u32x2;

static __device__ __forceinline__ float bf2f(unsigned short u) {
    return __uint_as_float(((unsigned)u) << 16);
}
static __device__ __forceinline__ unsigned short f2bf(float f) {
    unsigned u = __float_as_uint(f);
    return (unsigned short)((u + 0x7FFFu + ((u >> 16) & 1u)) >> 16);
}

union Frag32 {             // 32 bytes: two b128 LDS loads -> one WMMA operand
    u32x4 q[2];
    v16bf v;
};

#define TAPS 81
#define AROW 40            // sA row stride in shorts (80 B): bank-destaggered
#define BROW 40            // sBt row stride in shorts

// ---------------------------------------------------------------------------
// Implicit-GEMM 4D conv, bf16 WMMA, channels-last activations.
//   in : [Ti,Zi,Yi,Xi,Cin]   bf16 bits, Cin % 16 == 0 (input zero-padded)
//   wt : [81, Cout, Cin]     bf16 bits (pre-transformed, K-contiguous)
//   out: [To,Zo,Yo,Xo,Cout]  bf16 bits
// Block = 256 threads = 8 waves covering 128 M x 32 N.
// Each wave: 16(M) x 32(N) tile = two accumulators, A fragment reused.
// K loop: 81 taps x ceil(Cin/32): 2x v_wmma_f32_16x16x32_bf16 per chunk.
// Single fully-branchless staging path; all indexing 32-bit.
// ---------------------------------------------------------------------------
__global__ __launch_bounds__(256) void conv4d_wmma_k(
    const unsigned short* __restrict__ in,
    const unsigned short* __restrict__ wt,
    const float* __restrict__ bias,
    unsigned short* __restrict__ out,
    int Ti, int Zi, int Yi, int Xi, int Cin,
    int To, int Zo, int Yo, int Xo, int Cout,
    int stride, int relu)
{
    __shared__ alignas(16) unsigned short sA[128 * AROW];  // 10 KB
    __shared__ alignas(16) unsigned short sBt[32 * BROW];  // 2.5 KB (n-major)

    const int tid  = threadIdx.x;
    const int lane = tid & 31;
    const int wave = tid >> 5;
    const int M = To * Zo * Yo * Xo;
    const int mBlock = (int)blockIdx.x * 128;
    const int nBase  = (int)blockIdx.y * 32;

    // A-staging assignment: two threads per position row (16 channels each).
    const int aRow  = tid >> 1;              // 0..127
    const int aHalf = tid & 1;               // channel half 0..15 / 16..31
    const bool posOk = (mBlock + aRow) < M;
    int xo, yo, zo, to;
    {
        unsigned p = (unsigned)(posOk ? (mBlock + aRow) : 0);
        xo = (int)(p % (unsigned)Xo); p /= (unsigned)Xo;
        yo = (int)(p % (unsigned)Yo); p /= (unsigned)Yo;
        zo = (int)(p % (unsigned)Zo); p /= (unsigned)Zo;
        to = (int)p;
    }
    unsigned short* const aDst = &sA[aRow * AROW + aHalf * 16];

    // B-staging assignment: one (cout row, k-quad) per thread (b64).
    const int bN   = tid >> 3;               // 0..31  cout within tile
    const int bK   = (tid & 7) * 4;          // 0..28  k-quad within chunk
    const int bCo  = nBase + bN;
    const int bCoC = (bCo < Cout) ? bCo : (Cout - 1);
    unsigned short* const bDst = &sBt[bN * BROW + bK];

    // Fragment read pointers (fixed per thread).
    const u32x4* const aFrag  = (const u32x4*)&sA[(wave * 16 + (lane & 15)) * AROW];
    const u32x4* const bFrag0 = (const u32x4*)&sBt[(lane & 15) * BROW];
    const u32x4* const bFrag1 = (const u32x4*)&sBt[((lane & 15) + 16) * BROW];
    const int hA = (lane < 16) ? 0 : 2;      // A: k 0..15 / 16..31
    const int hB = (lane < 16) ? 0 : 2;      // B: k 0..15 / 16..31

    v8f acc[2];
    acc[0] = (v8f){0.f,0.f,0.f,0.f,0.f,0.f,0.f,0.f};
    acc[1] = (v8f){0.f,0.f,0.f,0.f,0.f,0.f,0.f,0.f};

    #pragma clang loop unroll(disable)
    for (int tap = 0; tap < TAPS; ++tap) {
        const int kt = tap / 27;
        int kr = tap % 27;
        const int kz = kr / 9; kr %= 9;
        const int ky = kr / 3;
        const int kx = kr % 3;

        const int ti = to * stride + kt - 1;
        const int zi = zo * stride + kz - 1;
        const int yi = yo * stride + ky - 1;
        const int xi = xo * stride + kx - 1;
        const bool inb = posOk &&
            (unsigned)ti < (unsigned)Ti && (unsigned)zi < (unsigned)Zi &&
            (unsigned)yi < (unsigned)Yi && (unsigned)xi < (unsigned)Xi;
        // Clamped base: 0 when OOB -> loads always legal, masked later.
        const int base = inb ? ((((ti * Zi + zi) * Yi + yi) * Xi + xi) * Cin) : 0;

        // Prefetch next tap's weights (gfx1250 global_prefetch path).
        if (tap + 1 < TAPS)
            __builtin_prefetch(&wt[(tap + 1) * Cout * Cin], 0, 1);

        const int wRow = (tap * Cout + bCoC) * Cin;

        #pragma clang loop unroll(disable)
        for (int k0 = 0; k0 < Cin; k0 += 32) {
            // ---- stage A: 16 channels per thread, branchless b128 ----
            {
                const int  kOff = k0 + aHalf * 16;
                const bool aOk  = inb && (kOff < Cin);     // Cin % 16 == 0
                const int  aoff = (kOff < Cin) ? (base + kOff) : 0;
                const u32x4* src = (const u32x4*)(in + aoff);
                u32x4 v0 = src[0];
                u32x4 v1 = src[1];
                if (!aOk) { u32x4 z = {0u,0u,0u,0u}; v0 = z; v1 = z; }
                ((u32x4*)aDst)[0] = v0;
                ((u32x4*)aDst)[1] = v1;
            }
            // ---- stage B: one b64 (4 k) per thread, branchless ----
            {
                const int  ci  = k0 + bK;
                const bool bOk = (bCo < Cout) && (ci < Cin); // b64 never split
                const int  boff = bOk ? (wRow + ci) : 0;
                u32x2 v = *(const u32x2*)(wt + boff);
                if (!bOk) { u32x2 z = {0u,0u}; v = z; }
                *(u32x2*)bDst = v;
            }
            __syncthreads();

            // ---- fragments: A reused across both N-halves ----
            Frag32 fa, fb0, fb1;
            fa.q[0]  = aFrag[hA];
            fa.q[1]  = aFrag[hA + 1];
            fb0.q[0] = bFrag0[hB];
            fb0.q[1] = bFrag0[hB + 1];
            fb1.q[0] = bFrag1[hB];
            fb1.q[1] = bFrag1[hB + 1];
            acc[0] = __builtin_amdgcn_wmma_f32_16x16x32_bf16(
                false, fa.v, false, fb0.v, (short)0, acc[0], false, false);
            acc[1] = __builtin_amdgcn_wmma_f32_16x16x32_bf16(
                false, fa.v, false, fb1.v, (short)0, acc[1], false, false);
            __syncthreads();
        }
    }

    // ---- epilogue: bias (+ReLU), bf16 store ----
    #pragma unroll
    for (int half = 0; half < 2; ++half) {
        const int co = nBase + (lane & 15) + half * 16;
        const float bv = (co < Cout) ? bias[co] : 0.f;
        #pragma unroll
        for (int r = 0; r < 8; ++r) {
            const int m = (lane < 16) ? r : (r + 8);
            const int pos = mBlock + wave * 16 + m;
            if (pos < M && co < Cout) {
                float v = acc[half][r] + bv;
                if (relu) v = fmaxf(v, 0.f);
                out[pos * Cout + co] = f2bf(v);
            }
        }
    }
}

// ---------------------------------------------------------------------------
// Weight prep: fp32 [Cout,CinS,3,3,3,3] -> bf16 [81,Cout,CinP] (K contiguous)
// Channels ci >= CinS are zero (layer-0 padding 1 -> 16).
// ---------------------------------------------------------------------------
__global__ void wprep_k(const float* __restrict__ src,
                        unsigned short* __restrict__ dst,
                        int CinS, int CinP, int Cout)
{
    const unsigned N = 81u * (unsigned)CinP * (unsigned)Cout;
    unsigned i = blockIdx.x * blockDim.x + threadIdx.x;
    if (i >= N) return;
    unsigned r = i;
    const unsigned ci  = r % (unsigned)CinP; r /= (unsigned)CinP;
    const unsigned co  = r % (unsigned)Cout; r /= (unsigned)Cout;
    const unsigned tap = r;
    dst[i] = (ci < (unsigned)CinS)
        ? f2bf(src[(co * (unsigned)CinS + ci) * 81u + tap])
        : (unsigned short)0;
}

// Input: fp32 [P] (C==1) -> bf16 channels-last [P,16], channel 0 = value.
__global__ void padin_k(const float* __restrict__ src,
                        unsigned short* __restrict__ dst, unsigned P)
{
    unsigned i = blockIdx.x * blockDim.x + threadIdx.x;
    if (i >= P * 16u) return;
    const unsigned p = i >> 4, c = i & 15u;
    dst[i] = (c == 0u) ? f2bf(src[p]) : (unsigned short)0;
}

// ---------------------------------------------------------------------------
// BatchNorm (no affine, no eps, matching reference): per-channel stats
// ---------------------------------------------------------------------------
__global__ void bn_stats_k(const unsigned short* __restrict__ x, int P, int C,
                           float* __restrict__ mean, float* __restrict__ rstd)
{
    __shared__ float s1[256], s2[256];
    const int c = blockIdx.x;
    float sum = 0.f, sq = 0.f;
    for (int p = threadIdx.x; p < P; p += blockDim.x) {
        const float v = bf2f(x[(unsigned)p * (unsigned)C + (unsigned)c]);
        sum += v; sq += v * v;
    }
    s1[threadIdx.x] = sum; s2[threadIdx.x] = sq;
    __syncthreads();
    for (int s = 128; s > 0; s >>= 1) {
        if (threadIdx.x < s) {
            s1[threadIdx.x] += s1[threadIdx.x + s];
            s2[threadIdx.x] += s2[threadIdx.x + s];
        }
        __syncthreads();
    }
    if (threadIdx.x == 0) {
        const float m = s1[0] / (float)P;
        const float v = s2[0] / (float)P - m * m;
        mean[c] = m;
        rstd[c] = rsqrtf(fmaxf(v, 1e-30f));
    }
}

__global__ void bn_apply_k(unsigned short* __restrict__ buf, unsigned N, int C,
                           const float* __restrict__ mean,
                           const float* __restrict__ rstd, int relu)
{
    unsigned i = blockIdx.x * blockDim.x + threadIdx.x;
    if (i >= N) return;
    const unsigned c = i % (unsigned)C;
    float v = (bf2f(buf[i]) - mean[c]) * rstd[c];
    if (relu) v = fmaxf(v, 0.f);
    buf[i] = f2bf(v);
}

// ---------------------------------------------------------------------------
// up2: repeat x2 along T,Z,Y,X (channels-last)
// ---------------------------------------------------------------------------
__global__ void up2_k(const unsigned short* __restrict__ in,
                      unsigned short* __restrict__ out,
                      int T, int Z, int Y, int X, int C)
{
    const unsigned N = (unsigned)T * Z * Y * X * C * 16u;
    unsigned i = blockIdx.x * blockDim.x + threadIdx.x;
    if (i >= N) return;
    unsigned r = i;
    const unsigned c = r % (unsigned)C;       r /= (unsigned)C;
    const unsigned x = r % (unsigned)(2 * X); r /= (unsigned)(2 * X);
    const unsigned y = r % (unsigned)(2 * Y); r /= (unsigned)(2 * Y);
    const unsigned z = r % (unsigned)(2 * Z); r /= (unsigned)(2 * Z);
    const unsigned t = r;
    out[i] = in[(((((t >> 1) * Z + (z >> 1)) * Y + (y >> 1)) * X + (x >> 1)) * C) + c];
}

// ---------------------------------------------------------------------------
// cat_pad: out[...,0:C1]=bypass, out[...,C1:]=zero-padded upsampled tensor
// ---------------------------------------------------------------------------
__global__ void catpad_k(const unsigned short* __restrict__ byp,
                         const unsigned short* __restrict__ ups,
                         unsigned short* __restrict__ out,
                         int T, int Z, int Y, int X, int C1, int C2,
                         int T2, int Z2, int Y2, int X2,
                         int pt, int pz, int py, int px)
{
    const int C = C1 + C2;
    const unsigned N = (unsigned)T * Z * Y * X * C;
    unsigned i = blockIdx.x * blockDim.x + threadIdx.x;
    if (i >= N) return;
    unsigned r = i;
    const int c = (int)(r % (unsigned)C); r /= (unsigned)C;
    const int x = (int)(r % (unsigned)X); r /= (unsigned)X;
    const int y = (int)(r % (unsigned)Y); r /= (unsigned)Y;
    const int z = (int)(r % (unsigned)Z); r /= (unsigned)Z;
    const int t = (int)r;
    if (c < C1) {
        out[i] = byp[(((t * Z + z) * Y + y) * X + x) * C1 + c];
    } else {
        const int t2 = t - pt, z2 = z - pz, y2 = y - py, x2 = x - px;
        unsigned short v = 0;
        if ((unsigned)t2 < (unsigned)T2 && (unsigned)z2 < (unsigned)Z2 &&
            (unsigned)y2 < (unsigned)Y2 && (unsigned)x2 < (unsigned)X2)
            v = ups[(((t2 * Z2 + z2) * Y2 + y2) * X2 + x2) * C2 + (c - C1)];
        out[i] = v;
    }
}

// ---------------------------------------------------------------------------
// 2-channel softmax + channels-last -> channel-major fp32 output
// ---------------------------------------------------------------------------
__global__ void softmax2_k(const unsigned short* __restrict__ in,
                           float* __restrict__ out, unsigned P)
{
    unsigned p = blockIdx.x * blockDim.x + threadIdx.x;
    if (p >= P) return;
    const float a = bf2f(in[p * 2u + 0u]);
    const float b = bf2f(in[p * 2u + 1u]);
    const float m = fmaxf(a, b);
    const float ea = __expf(a - m), eb = __expf(b - m);
    const float s = ea + eb;
    out[p]     = ea / s;
    out[P + p] = eb / s;
}

// ---------------------------------------------------------------------------
// Host orchestration
// ---------------------------------------------------------------------------
struct Dims { int T, Z, Y, X, C; };

extern "C" void kernel_launch(void* const* d_in, const int* in_sizes, int n_in,
                              void* d_out, int out_size, void* d_ws, size_t ws_size,
                              hipStream_t stream)
{
    (void)in_sizes; (void)n_in; (void)out_size; (void)ws_size;

    static const int CIN[23]  = {1,16,32,32,32,64,64,64,128,128,128,256,256,256,
                                 768,256,384,128,192,64,96,32,32};
    static const int COUT[23] = {16,32,32,32,64,64,64,128,128,128,256,256,256,512,
                                 256,256,128,128,64,64,32,32,2};
    // Padded Cin (layer 0: 1 -> 16) so the conv kernel sees Cin % 16 == 0.
    static const int CINP[23] = {16,16,32,32,32,64,64,64,128,128,128,256,256,256,
                                 768,256,384,128,192,64,96,32,32};

    const float* x = (const float*)d_in[0];
    const float* W[23]; const float* Bs[23];
    for (int i = 0; i < 23; ++i) {
        W[i]  = (const float*)d_in[1 + 2 * i];
        Bs[i] = (const float*)d_in[2 + 2 * i];
    }

    char* ws = (char*)d_ws;
    size_t off = 0;
    auto alloc = [&](size_t bytes) -> char* {
        char* p = ws + off;
        off = (off + bytes + 255) & ~(size_t)255;
        return p;
    };

    // ---- weights -> bf16 [81,Cout,CinP] ----
    unsigned short* wbf[23];
    for (int i = 0; i < 23; ++i) {
        const unsigned n = 81u * (unsigned)CINP[i] * (unsigned)COUT[i];
        wbf[i] = (unsigned short*)alloc((size_t)n * 2);
        wprep_k<<<(n + 255) / 256, 256, 0, stream>>>(W[i], wbf[i], CIN[i], CINP[i], COUT[i]);
    }

    float* stats = (float*)alloc(1024 * sizeof(float));   // mean[512] + rstd[512]

    // ---- input -> bf16 channels-last, padded to 16 channels ----
    Dims dx{16, 16, 32, 32, 16};
    const unsigned Px = (unsigned)dx.T * dx.Z * dx.Y * dx.X;
    unsigned short* xb = (unsigned short*)alloc((size_t)Px * 16 * 2);
    padin_k<<<(Px * 16 + 255) / 256, 256, 0, stream>>>(x, xb, Px);

    auto conv = [&](const unsigned short* in, Dims din, int li, int stride,
                    int relu, Dims& dout) -> unsigned short* {
        dout.T = (din.T - 1) / stride + 1;   // (T + 2*1 - 3)/stride + 1
        dout.Z = (din.Z - 1) / stride + 1;
        dout.Y = (din.Y - 1) / stride + 1;
        dout.X = (din.X - 1) / stride + 1;
        dout.C = COUT[li];
        const unsigned M = (unsigned)dout.T * dout.Z * dout.Y * dout.X;
        unsigned short* out = (unsigned short*)alloc((size_t)M * dout.C * 2);
        dim3 grid((M + 127) / 128, (unsigned)((COUT[li] + 31) / 32));
        conv4d_wmma_k<<<grid, 256, 0, stream>>>(
            in, wbf[li], Bs[li], out,
            din.T, din.Z, din.Y, din.X, din.C,
            dout.T, dout.Z, dout.Y, dout.X, dout.C, stride, relu);
        return out;
    };

    auto bn = [&](unsigned short* buf, Dims d, int relu) {
        const unsigned Pn = (unsigned)d.T * d.Z * d.Y * d.X;
        bn_stats_k<<<d.C, 256, 0, stream>>>(buf, (int)Pn, d.C, stats, stats + 512);
        const unsigned N = Pn * (unsigned)d.C;
        bn_apply_k<<<(N + 255) / 256, 256, 0, stream>>>(
            buf, N, d.C, stats, stats + 512, relu);
    };

    auto up = [&](const unsigned short* in, Dims din, Dims& dout) -> unsigned short* {
        dout = Dims{din.T * 2, din.Z * 2, din.Y * 2, din.X * 2, din.C};
        const unsigned N = (unsigned)dout.T * dout.Z * dout.Y * dout.X * dout.C;
        unsigned short* out = (unsigned short*)alloc((size_t)N * 2);
        up2_k<<<(N + 255) / 256, 256, 0, stream>>>(
            in, out, din.T, din.Z, din.Y, din.X, din.C);
        return out;
    };

    auto catpad = [&](const unsigned short* byp, Dims db,
                      const unsigned short* ups, Dims du, Dims& dout) -> unsigned short* {
        dout = Dims{db.T, db.Z, db.Y, db.X, db.C + du.C};
        const unsigned N = (unsigned)dout.T * dout.Z * dout.Y * dout.X * dout.C;
        unsigned short* out = (unsigned short*)alloc((size_t)N * 2);
        catpad_k<<<(N + 255) / 256, 256, 0, stream>>>(
            byp, ups, out, db.T, db.Z, db.Y, db.X, db.C, du.C,
            du.T, du.Z, du.Y, du.X,
            (db.T - du.T) / 2, (db.Z - du.Z) / 2, (db.Y - du.Y) / 2, (db.X - du.X) / 2);
        return out;
    };

    // ---- encoder ----
    Dims t, de1, dp1, de2, dp2, de3, dp3, de4, dp4;
    unsigned short* c  = conv(xb, dx, 0, 1, 1, t);
    unsigned short* e1 = conv(c, t, 1, 1, 0, de1);  bn(e1, de1, 1);
    unsigned short* q1 = conv(e1, de1, 2, 2, 0, dp1);
    c = conv(q1, dp1, 3, 1, 1, t);
    unsigned short* e2 = conv(c, t, 4, 1, 0, de2);  bn(e2, de2, 1);
    unsigned short* q2 = conv(e2, de2, 5, 2, 0, dp2);
    c = conv(q2, dp2, 6, 1, 1, t);
    unsigned short* e3 = conv(c, t, 7, 1, 0, de3);  bn(e3, de3, 1);
    unsigned short* q3 = conv(e3, de3, 8, 2, 0, dp3);
    c = conv(q3, dp3, 9, 1, 1, t);
    unsigned short* e4 = conv(c, t, 10, 1, 0, de4); bn(e4, de4, 1);
    unsigned short* q4 = conv(e4, de4, 11, 2, 0, dp4);

    // ---- bottleneck ----
    Dims db1, db2, db2u;
    unsigned short* b1  = conv(q4, dp4, 12, 1, 0, db1);  bn(b1, db1, 1);
    unsigned short* b2p = conv(b1, db1, 13, 1, 0, db2);  bn(b2p, db2, 1);
    unsigned short* b2  = up(b2p, db2, db2u);

    // ---- decoder ----
    Dims dd4, du4, du4u, dd3, du3, du3u, dd2, du2, du2u, dd1;
    unsigned short* d4  = catpad(e4, de4, b2, db2u, dd4);
    c = conv(d4, dd4, 14, 1, 1, t);
    unsigned short* u4p = conv(c, t, 15, 1, 0, du4);  bn(u4p, du4, 1);
    unsigned short* u4  = up(u4p, du4, du4u);

    unsigned short* d3  = catpad(e3, de3, u4, du4u, dd3);
    c = conv(d3, dd3, 16, 1, 1, t);
    unsigned short* u3p = conv(c, t, 17, 1, 0, du3);  bn(u3p, du3, 1);
    unsigned short* u3  = up(u3p, du3, du3u);

    unsigned short* d2  = catpad(e2, de2, u3, du3u, dd2);
    c = conv(d2, dd2, 18, 1, 1, t);
    unsigned short* u2p = conv(c, t, 19, 1, 0, du2);  bn(u2p, du2, 1);
    unsigned short* u2  = up(u2p, du2, du2u);

    unsigned short* d1  = catpad(e1, de1, u2, du2u, dd1);

    // ---- head ----
    Dims df1, df2, df3;
    unsigned short* f = conv(d1, dd1, 20, 1, 0, df1); bn(f, df1, 1);
    f = conv(f, df1, 21, 1, 0, df2);                  bn(f, df2, 1);
    f = conv(f, df2, 22, 1, 0, df3);                  bn(f, df3, 0);

    const unsigned Pf = (unsigned)df3.T * df3.Z * df3.Y * df3.X;
    softmax2_k<<<(Pf + 255) / 256, 256, 0, stream>>>(f, (float*)d_out, Pf);
}
